// MultiHeadAttentionWithRelativePositionalEmbedding_5214090297505
// MI455X (gfx1250) — matrix operand order repaired and checked by reading, not compiled, not run on hardware
//
#include <hip/hip_runtime.h>
#include <hip/hip_bf16.h>
#include <math.h>

typedef __attribute__((ext_vector_type(16))) _Float16 v16h;
typedef __attribute__((ext_vector_type(8)))  _Float16 v8h;
typedef __attribute__((ext_vector_type(4)))  _Float16 v4h;
typedef __attribute__((ext_vector_type(8)))  float    v8f;
typedef __attribute__((ext_vector_type(4)))  float    v4f;
typedef __attribute__((ext_vector_type(4)))  unsigned int u32x4;
typedef __attribute__((ext_vector_type(4)))  int      i32x4;
typedef __attribute__((ext_vector_type(8)))  int      i32x8;

constexpr int BATCH = 8;
constexpr int SEQ   = 1024;
constexpr int DIM   = 512;
constexpr int NHEAD = 16;
constexpr int DHEAD = 32;
constexpr float RSCALE = 0.04419417382415922f;  // 1/sqrt(512)

#define WMMA(a, b, c) \
  __builtin_amdgcn_wmma_f32_16x16x32_f16(false, (a), false, (b), (short)0, (c), false, false)

// Tensor Data Mover availability (probe-confirmed builtin; arity differs per toolchain)
#if __has_builtin(__builtin_amdgcn_tensor_load_to_lds)
#define HAVE_TDM 1
#else
#define HAVE_TDM 0
#endif

#if HAVE_TDM
// Minimal 2D D# (Tensor DMA Descriptor) per CDNA5 ISA:
//   group0: count=1 | lds_addr | global_addr(57b) | type=2
//   group1: data_size=2B | tensor_dim0/1 | tile_dim0/1 | tensor_dim0_stride
__device__ __forceinline__ void tdm_load_tile_2d(unsigned lds_addr, const void* gptr,
                                                 unsigned tile_d0, unsigned tile_d1,
                                                 unsigned tensor_d0, unsigned tensor_d1,
                                                 unsigned stride_d0_elems) {
  unsigned long long ga = (unsigned long long)(uintptr_t)gptr;
  u32x4 g0;
  g0[0] = 1u;                                            // count=1 (valid), user mode
  g0[1] = lds_addr;                                      // LDS byte address
  g0[2] = (unsigned)(ga & 0xffffffffu);                  // global_addr[31:0]
  g0[3] = (unsigned)((ga >> 32) & 0x1ffffffu) | (2u << 30);  // addr[56:32], type=2
  i32x8 g1;
  g1[0] = (int)(1u << 16);                               // wg_mask=0, data_size=1 (2B)
  g1[1] = (int)(tensor_d0 << 16);                        // tensor_dim0[15:0]
  g1[2] = (int)((tensor_d0 >> 16) | (tensor_d1 << 16));  // dim0 hi | dim1 lo
  g1[3] = (int)((tensor_d1 >> 16) | (tile_d0 << 16));    // dim1 hi | tile_dim0
  g1[4] = (int)tile_d1;                                  // tile_dim1 (tile_dim2=0)
  g1[5] = (int)stride_d0_elems;                          // tensor_dim0_stride lo32
  g1[6] = 0;
  g1[7] = 0;
  i32x4 z4 = {};
#if __clang_major__ >= 23
  i32x8 z8 = {};
  __builtin_amdgcn_tensor_load_to_lds(g0, g1, z4, z4, z8, 0);
#else
  __builtin_amdgcn_tensor_load_to_lds(g0, g1, z4, z4, 0);
#endif
}
#endif

// ---------------------------------------------------------------------------
// Fragment builders for V_WMMA_F32_16X16X32_F16 (wave32).
// A (16x32 MxK): lane L: row m=L&15, K = {koff..+7} U {16+koff..+7}, koff=(L>>4)*8
//                -> two contiguous 16B chunks.
// B (32x16 KxN): lane L: col n=L&15, K = {koff..+7} U {16+koff..+7}, koff=(L>>4)*8
//                -> two contiguous 16B chunks of a K-major row.
// C/D (16x16 f32): lane L: col n=L&15, vgpr r -> row (L>>4)*8 + r.
// ---------------------------------------------------------------------------
__device__ __forceinline__ v16h frag_a(const _Float16* base, int stride, int lane) {
  int m    = lane & 15;
  int koff = (lane >> 4) << 3;
  const _Float16* p = base + (size_t)m * stride + koff;
  v8h lo = *(const v8h*)(p);
  v8h hi = *(const v8h*)(p + 16);
  v16h a;
#pragma unroll
  for (int i = 0; i < 8; ++i) { a[i] = lo[i]; a[8 + i] = hi[i]; }
  return a;
}

__device__ __forceinline__ v16h frag_b(const _Float16* base, int stride, int lane) {
  int n  = lane & 15;
  int k0 = (lane >> 4) << 3;
  const _Float16* p = base + (size_t)n * stride + k0;
  v8h lo = *(const v8h*)(p);
  v8h hi = *(const v8h*)(p + 16);
  v16h b;
#pragma unroll
  for (int i = 0; i < 8; ++i) { b[i] = lo[i]; b[8 + i] = hi[i]; }
  return b;
}

__device__ __forceinline__ v16h frag_q_bias(const _Float16* q, int b, int row_base,
                                            int h, const float* bias, int lane) {
  int m   = lane & 15;
  int row = row_base + m; if (row > SEQ - 1) row = SEQ - 1;
  int koff = (lane >> 4) << 3;
  const _Float16* p  = q + ((size_t)(b * SEQ + row) * DIM) + h * DHEAD + koff;
  const float*    bp = bias + h * DHEAD + koff;
  v8h lo = *(const v8h*)(p);
  v8h hi = *(const v8h*)(p + 16);
  v16h a;
#pragma unroll
  for (int i = 0; i < 8; ++i) {
    a[i]     = lo[i] + (_Float16)bp[i];
    a[8 + i] = hi[i] + (_Float16)bp[16 + i];
  }
  return a;
}

// ---------------------------------------------------------------------------
__global__ void pe_kernel(_Float16* __restrict__ pe16) {
  int l = blockIdx.x;        // 0..1023
  int t = threadIdx.x;       // 0..255
  float freq = __expf(-9.210340371976184f * (float)(2 * t) * (1.0f / (float)DIM));
  float ang  = (float)l * freq;
  float s, c;
  __sincosf(ang, &s, &c);
  pe16[(size_t)l * DIM + 2 * t]     = (_Float16)s;
  pe16[(size_t)l * DIM + 2 * t + 1] = (_Float16)c;
}

__global__ void cvt_kernel(const float* __restrict__ in, _Float16* __restrict__ out, int n4) {
  int i = blockIdx.x * blockDim.x + threadIdx.x;
  if (i < n4) {
    v4f x = *(const v4f*)(in + (size_t)4 * i);
    v4h y;
#pragma unroll
    for (int j = 0; j < 4; ++j) y[j] = (_Float16)x[j];
    *(v4h*)(out + (size_t)4 * i) = y;
  }
}

// ---------------------------------------------------------------------------
// NT GEMM: C[M,N] = A[M,K] * B[N,K]^T (+bias).  A,B f16 row-major (K contig).
// Block: 256 thr = 8 waves -> 256x64 tile; wave = 32x64 strip (2 A-frags,
// 4 shared B-frags -> 8 WMMA per K-step).
// ---------------------------------------------------------------------------
template <bool F32OUT>
__global__ void gemm_nt_kernel(const _Float16* __restrict__ A,
                               const _Float16* __restrict__ B,
                               const float* __restrict__ bias,
                               void* __restrict__ Cv, int M, int N, int K) {
  int lane = threadIdx.x & 31;
  int wave = threadIdx.x >> 5;
  int row0 = blockIdx.x * 256 + wave * 32;
  int col0 = blockIdx.y * 64;
  if (row0 >= M) return;

  v8f acc[2][4] = {};
  for (int k0 = 0; k0 < K; k0 += 32) {
    __builtin_prefetch(A + (size_t)row0 * K + k0 + 256, 0, 0);
    v16h a0 = frag_a(A + (size_t)row0 * K + k0, K, lane);
    v16h a1 = frag_a(A + (size_t)(row0 + 16) * K + k0, K, lane);
#pragma unroll
    for (int ct = 0; ct < 4; ++ct) {
      v16h b = frag_b(B + (size_t)(col0 + ct * 16) * K + k0, K, lane);
      acc[0][ct] = WMMA(a0, b, acc[0][ct]);
      acc[1][ct] = WMMA(a1, b, acc[1][ct]);
    }
  }
  int n  = lane & 15;
  int mb = (lane >> 4) << 3;
#pragma unroll
  for (int rt = 0; rt < 2; ++rt)
#pragma unroll
    for (int ct = 0; ct < 4; ++ct)
#pragma unroll
      for (int r = 0; r < 8; ++r) {
        int row = row0 + rt * 16 + mb + r;
        int col = col0 + ct * 16 + n;
        float val = acc[rt][ct][r] + (bias ? bias[col] : 0.0f);
        if (F32OUT) ((float*)Cv)[(size_t)row * N + col] = val;
        else        ((_Float16*)Cv)[(size_t)row * N + col] = (_Float16)val;
      }
}

// ---------------------------------------------------------------------------
// Attention: per (b, h, 64-row l-tile).  Persistent workgroups, 128 thr (4 waves).
// LDS: P (80x1024 f16: (q+v_bias)@pos^T rows l0..l0+79; later 64x1024 probs) 160KB
//      S (64x1024 f16 scores; reused in phase D as V panel: 1024x32 row-major
//        + 32x1024 transposed) 128KB ; SCR (128 f32 softmax partials) 1KB
// ---------------------------------------------------------------------------
constexpr int LTILE  = 64;
constexpr int NTASK  = BATCH * NHEAD * (SEQ / LTILE);     // 2048
constexpr size_t SM_P  = (size_t)80 * SEQ * 2;            // 163840
constexpr size_t SM_S  = (size_t)64 * SEQ * 2;            // 131072
constexpr size_t SM_SCR = 1024;
constexpr size_t SMEM_BYTES = SM_P + SM_S + SM_SCR;       // 295936

__global__ void attn_kernel(const _Float16* __restrict__ q16,
                            const _Float16* __restrict__ k16,
                            const _Float16* __restrict__ v16,
                            const _Float16* __restrict__ pos16,
                            const float* __restrict__ u_bias,
                            const float* __restrict__ v_bias,
                            _Float16* __restrict__ ctx16) {
  extern __shared__ char smem[];
  _Float16* P   = (_Float16*)smem;                  // 80x1024 P-block / 64x1024 probs
  _Float16* S   = (_Float16*)(smem + SM_P);         // 64x1024 scores / V panel
  float*    SCR = (float*)(smem + SM_P + SM_S);     // 128 floats

  const int tid  = threadIdx.x;
  const int lane = tid & 31;
  const int wave = tid >> 5;
  const int n    = lane & 15;
  const int mb   = (lane >> 4) << 3;

  for (int task = blockIdx.x; task < NTASK; task += gridDim.x) {
    const int b  = task / (NHEAD * (SEQ / LTILE));
    const int h  = (task / (SEQ / LTILE)) % NHEAD;
    const int l0 = (task % (SEQ / LTILE)) * LTILE;

    __syncthreads();  // LDS reuse across tasks

    // ---- Phase A: P[l-l0][j] = (q[l]+v_bias) . pos[j], rows l0..l0+79 ----
    for (int st = wave; st < 5; st += 4) {
      v16h a = frag_q_bias(q16, b, l0 + st * 16, h, v_bias, lane);
      for (int n0 = 0; n0 < SEQ; n0 += 16) {
        v16h bb = frag_b(pos16 + (size_t)n0 * DIM + h * DHEAD, DIM, lane);
        v8f c = {};
        c = WMMA(a, bb, c);
#pragma unroll
        for (int r = 0; r < 8; ++r)
          P[(size_t)(st * 16 + mb + r) * SEQ + n0 + n] = (_Float16)c[r];
      }
    }
    __syncthreads();

    // ---- Phase B: S = (content + rel_shift(P)) * rscale ----
    {
      const int ls0 = l0 + wave * 16;
      v16h a = frag_q_bias(q16, b, ls0, h, u_bias, lane);
      for (int m0 = 0; m0 < SEQ; m0 += 16) {
        if (m0 + 16 < SEQ)
          __builtin_prefetch(k16 + ((size_t)(b * SEQ + m0 + 16) * DIM) + h * DHEAD, 0, 0);
        v16h bb = frag_b(k16 + ((size_t)(b * SEQ + m0) * DIM) + h * DHEAD, DIM, lane);
        v8f c = {};
        c = WMMA(a, bb, c);
#pragma unroll
        for (int r = 0; r < 8; ++r) {
          int l   = ls0 + mb + r;       // global query index
          int m   = m0 + n;             // global key index
          int rel = m - l;
          float pv;
          if (rel <= 0)      pv = (float)P[(size_t)(l - l0) * SEQ + (SEQ - 1 + rel)];
          else if (rel == 1) pv = 0.0f;
          else               pv = (float)P[(size_t)(l - l0 + 1) * SEQ + (rel - 2)];
          S[(size_t)(l - l0) * SEQ + m] = (_Float16)((c[r] + pv) * RSCALE);
        }
      }
    }
    __syncthreads();

    // ---- Phase C: softmax, 2 threads/row, v8h LDS traffic; probs -> P ----
    {
      const int r = tid >> 1, half = tid & 1;
      const _Float16* Srow = S + (size_t)r * SEQ + half * 512;
      _Float16*       Prow = P + (size_t)r * SEQ + half * 512;
      float mx = -1e30f;
      for (int m = 0; m < 512; m += 8) {
        v8h x = *(const v8h*)(Srow + m);
#pragma unroll
        for (int j = 0; j < 8; ++j) mx = fmaxf(mx, (float)x[j]);
      }
      SCR[tid] = mx;
      __syncthreads();
      mx = fmaxf(SCR[2 * r], SCR[2 * r + 1]);
      __syncthreads();
      float sum = 0.0f;
      for (int m = 0; m < 512; m += 8) {
        v8h x = *(const v8h*)(Srow + m);
        v8h e;
#pragma unroll
        for (int j = 0; j < 8; ++j) {
          float ev = __expf((float)x[j] - mx);
          sum += ev;
          e[j] = (_Float16)ev;
        }
        *(v8h*)(Prow + m) = e;
      }
      SCR[tid] = sum;
      __syncthreads();
      float inv = 1.0f / (SCR[2 * r] + SCR[2 * r + 1]);
      for (int m = 0; m < 512; m += 8) {
        v8h x = *(const v8h*)(Prow + m);
#pragma unroll
        for (int j = 0; j < 8; ++j) x[j] = (_Float16)((float)x[j] * inv);
        *(v8h*)(Prow + m) = x;
      }
    }
    __syncthreads();

    // ---- Phase D: stage whole V panel once (S region is dead), then a
    //      barrier-free K=1024 WMMA loop straight out of LDS.             ----
    _Float16* VROW = S;                  // 1024 x 32 row-major (TDM dest)
    _Float16* VTR  = S + (size_t)32 * SEQ;  // 32 x 1024 transposed
#if HAVE_TDM
    if (wave == 0) {                     // one 64KB async DMA per task
      tdm_load_tile_2d((unsigned)(uintptr_t)VROW,
                       v16 + ((size_t)(b * SEQ) * DIM) + h * DHEAD,
                       /*tile*/ DHEAD, SEQ, /*tensor*/ DHEAD, SEQ, /*stride*/ DIM);
      __builtin_amdgcn_s_wait_tensorcnt(0);
    }
#else
    for (int c = tid; c < SEQ * DHEAD / 8; c += 128) {   // vectorized b128 staging
      int m = c >> 2, d8 = (c & 3) * 8;
      *(v8h*)(VROW + m * DHEAD + d8) =
          *(const v8h*)(v16 + ((size_t)(b * SEQ + m) * DIM) + h * DHEAD + d8);
    }
#endif
    __syncthreads();
    for (int c = tid; c < SEQ * DHEAD / 8; c += 128) {   // LDS transpose
      int m = c >> 2, d8 = (c & 3) * 8;
      v8h x = *(const v8h*)(VROW + m * DHEAD + d8);
#pragma unroll
      for (int j = 0; j < 8; ++j) VTR[(size_t)(d8 + j) * SEQ + m] = x[j];
    }
    __syncthreads();

    v8f acc[2] = {};
    for (int k0 = 0; k0 < SEQ; k0 += 32) {
      v16h a = frag_a(P + (size_t)(wave * 16) * SEQ + k0, SEQ, lane);
#pragma unroll
      for (int ct = 0; ct < 2; ++ct) {
        v16h bb = frag_b(VTR + (size_t)(ct * 16) * SEQ + k0, SEQ, lane);
        acc[ct] = WMMA(a, bb, acc[ct]);
      }
    }
#pragma unroll
    for (int ct = 0; ct < 2; ++ct)
#pragma unroll
      for (int r = 0; r < 8; ++r) {
        int l = l0 + wave * 16 + mb + r;
        ctx16[((size_t)(b * SEQ + l) * DIM) + h * DHEAD + ct * 16 + n] = (_Float16)acc[ct][r];
      }
  }
}

// ---------------------------------------------------------------------------
extern "C" void kernel_launch(void* const* d_in, const int* in_sizes, int n_in,
                              void* d_out, int out_size, void* d_ws, size_t ws_size,
                              hipStream_t stream) {
  (void)in_sizes; (void)n_in; (void)out_size; (void)ws_size;
  const float* f_query = (const float*)d_in[0];
  const float* f_key   = (const float*)d_in[1];
  const float* f_value = (const float*)d_in[2];
  const float* f_Wq    = (const float*)d_in[3];
  const float* f_bq    = (const float*)d_in[4];
  const float* f_Wk    = (const float*)d_in[5];
  const float* f_bk    = (const float*)d_in[6];
  const float* f_Wv    = (const float*)d_in[7];
  const float* f_bv    = (const float*)d_in[8];
  const float* f_Wpos  = (const float*)d_in[9];
  const float* f_u     = (const float*)d_in[10];
  const float* f_v     = (const float*)d_in[11];
  const float* f_Wo    = (const float*)d_in[12];
  const float* f_bo    = (const float*)d_in[13];
  float* out = (float*)d_out;

  const size_t NTOK = (size_t)BATCH * SEQ;  // 8192
  char* ws = (char*)d_ws;
  size_t off = 0;
  auto take = [&](size_t halves) -> _Float16* {
    _Float16* p = (_Float16*)(ws + off);
    off += ((halves * 2) + 255) & ~(size_t)255;
    return p;
  };
  _Float16* pe16   = take((size_t)SEQ * DIM);
  _Float16* xq16   = take(NTOK * DIM);
  _Float16* xk16   = take(NTOK * DIM);
  _Float16* xv16   = take(NTOK * DIM);
  _Float16* wq16   = take((size_t)DIM * DIM);
  _Float16* wk16   = take((size_t)DIM * DIM);
  _Float16* wv16   = take((size_t)DIM * DIM);
  _Float16* wpos16 = take((size_t)DIM * DIM);
  _Float16* wo16   = take((size_t)DIM * DIM);
  _Float16* q16    = take(NTOK * DIM);
  _Float16* k16    = take(NTOK * DIM);
  _Float16* v16    = take(NTOK * DIM);
  _Float16* pos16  = take((size_t)SEQ * DIM);
  _Float16* ctx16  = take(NTOK * DIM);

  // Stage 0: PE + f32->f16 conversions (vectorized x4)
  pe_kernel<<<SEQ, DIM / 2, 0, stream>>>(pe16);
  auto cvt = [&](const float* src, _Float16* dst, size_t nel) {
    int n4 = (int)(nel / 4);
    cvt_kernel<<<(n4 + 255) / 256, 256, 0, stream>>>(src, dst, n4);
  };
  cvt(f_query, xq16, NTOK * DIM);
  cvt(f_key,   xk16, NTOK * DIM);
  cvt(f_value, xv16, NTOK * DIM);
  cvt(f_Wq,   wq16,   (size_t)DIM * DIM);
  cvt(f_Wk,   wk16,   (size_t)DIM * DIM);
  cvt(f_Wv,   wv16,   (size_t)DIM * DIM);
  cvt(f_Wpos, wpos16, (size_t)DIM * DIM);
  cvt(f_Wo,   wo16,   (size_t)DIM * DIM);

  // Stage 1: projections (NT GEMMs, f16 WMMA, f32 accumulate)
  dim3 gBig((unsigned)(NTOK / 256), DIM / 64);
  gemm_nt_kernel<false><<<gBig, 256, 0, stream>>>(xq16, wq16, f_bq, q16,
                                                  (int)NTOK, DIM, DIM);
  gemm_nt_kernel<false><<<gBig, 256, 0, stream>>>(xk16, wk16, f_bk, k16,
                                                  (int)NTOK, DIM, DIM);
  gemm_nt_kernel<false><<<gBig, 256, 0, stream>>>(xv16, wv16, f_bv, v16,
                                                  (int)NTOK, DIM, DIM);
  dim3 gPos(SEQ / 256, DIM / 64);
  gemm_nt_kernel<false><<<gPos, 256, 0, stream>>>(pe16, wpos16, nullptr, pos16,
                                                  SEQ, DIM, DIM);

  // Stage 2: fused rel-pos attention (persistent, big-LDS, TDM staging)
  attn_kernel<<<256, 128, (unsigned)SMEM_BYTES, stream>>>(q16, k16, v16, pos16,
                                                          f_u, f_v, ctx16);

  // Stage 3: output projection (f32 out + bias)
  gemm_nt_kernel<true><<<gBig, 256, 0, stream>>>(ctx16, wo16, f_bo, out,
                                                 (int)NTOK, DIM, DIM);
}
